// MPNNELCell_32452772888814
// MI455X (gfx1250) — compile-verified, matrix operand order checked by reading, not run on hardware
//
#include <hip/hip_runtime.h>
#include <cmath>

// ---------------------------------------------------------------------------
// MPNN-EL cell forward for MI455X (gfx1250), wave32, WMMA bf16 path.
// All operands are pre-converted to bf16 exactly once; GEMM / attention inner
// loops are pure b128 loads + v_wmma_f32_16x16x32_bf16 (no conversion VALU).
// ---------------------------------------------------------------------------

#define TT   1024      // tokens (H*W)
#define BB   8         // batch
#define HC   512
#define MTOT (TT*BB)   // 8192 rows for token-wise linears
#define NHD  8
#define EPS  1e-5f

typedef __attribute__((ext_vector_type(16))) __bf16 v16bf;
typedef __attribute__((ext_vector_type(8)))  __bf16 v8bf;
typedef __attribute__((ext_vector_type(8)))  float  v8f;

__device__ __forceinline__ __bf16 f2bf(float f) {
  union { float f; unsigned u; } c; c.f = f;
  unsigned u = c.u + 0x7FFFu + ((c.u >> 16) & 1u);   // round-to-nearest-even
  unsigned short h = (unsigned short)(u >> 16);
  return __builtin_bit_cast(__bf16, h);
}

#define CAT16(a, b) __builtin_shufflevector(a, b, 0,1,2,3,4,5,6,7,8,9,10,11,12,13,14,15)

// activation codes
#define ACT_NONE 0
#define ACT_RELU 1
#define ACT_SIG  2
#define ACT_TANH 3
#define ACT_GELU 4

__device__ __forceinline__ float apply_act(float y, int act) {
  switch (act) {
    case ACT_RELU: return y > 0.f ? y : 0.f;
    case ACT_SIG:  return 1.f / (1.f + __expf(-y));
    case ACT_TANH: return tanhf(y);
    case ACT_GELU: return 0.5f * y * (1.f + erff(y * 0.70710678118654752f));
    default:       return y;
  }
}

// ---------------------------------------------------------------------------
// fp32 -> bf16 bulk conversion (one b128 store per 8 elements)
// ---------------------------------------------------------------------------
__global__ void k_cvt_bf16(const float* __restrict__ in, __bf16* __restrict__ out, size_t n8)
{
  const size_t tid = (size_t)blockIdx.x * blockDim.x + threadIdx.x;
  if (tid >= n8) return;
  const size_t base = tid * 8;
  v8bf v;
  #pragma unroll
  for (int e = 0; e < 8; ++e) v[e] = f2bf(in[base + e]);
  *(v8bf*)(out + base) = v;
}

// ---------------------------------------------------------------------------
// GEMM: out[M,N] = act( A[M,K](bf16) @ W[N,K](bf16)^T + bias (+ resid) )
// Block = 256 (8 waves), wave = 16x64 tile, grid = (M/128, N/64), K % 32 == 0.
// Fragment layout (documented): lane l, lo=l&15 row/col, kb=8*(l>>4);
// v16bf element e<8 -> K=kb+e ; e>=8 -> K=kb+16+(e-8)  => two 16B loads.
// f32 and bf16 outputs both optional (bf16 copy feeds downstream GEMMs).
// ---------------------------------------------------------------------------
__global__ __launch_bounds__(256)
void k_gemm(const __bf16* __restrict__ A, int lda,
            const __bf16* __restrict__ W, int ldw,
            const float* __restrict__ bias,
            const float* __restrict__ resid,
            float* __restrict__ out,
            __bf16* __restrict__ outb,
            int ldo, int K, int act)
{
  const int l  = threadIdx.x & 31;
  const int wv = threadIdx.x >> 5;
  const int lo = l & 15, hi = l >> 4, kb = hi * 8;
  const int m0 = blockIdx.x * 128 + wv * 16;
  const int n0 = blockIdx.y * 64;

  v8f acc[4];
  #pragma unroll
  for (int c = 0; c < 4; ++c) acc[c] = (v8f)(0.f);

  const __bf16* ap0 = A + (size_t)(m0 + lo) * lda + kb;
  const __bf16* wp0 = W + (size_t)(n0 + lo) * ldw + kb;

  for (int k0 = 0; k0 < K; k0 += 32) {
    const __bf16* ap = ap0 + k0;
    v16bf af = CAT16(*(const v8bf*)(ap), *(const v8bf*)(ap + 16));
    if (k0 + 32 < K) {
      __builtin_prefetch(ap + 32, 0, 1);            // global_prefetch_b8
      __builtin_prefetch(wp0 + k0 + 32, 0, 1);
    }
    #pragma unroll
    for (int c = 0; c < 4; ++c) {
      const __bf16* wp = wp0 + (size_t)c * 16 * ldw + k0;
      v16bf bf_ = CAT16(*(const v8bf*)(wp), *(const v8bf*)(wp + 16));
      acc[c] = __builtin_amdgcn_wmma_f32_16x16x32_bf16(
          false, af, false, bf_, (short)0, acc[c], false, false);
    }
  }

  #pragma unroll
  for (int c = 0; c < 4; ++c) {
    const int n = n0 + c * 16 + lo;
    const float bv = bias ? bias[n] : 0.f;
    #pragma unroll
    for (int i = 0; i < 8; ++i) {
      const int m = m0 + i + hi * 8;               // C layout: lane-half -> M block
      float y = acc[c][i] + bv;
      if (resid) y += resid[(size_t)m * ldo + n];
      y = apply_act(y, act);
      if (out)  out[(size_t)m * ldo + n]  = y;
      if (outb) outb[(size_t)m * ldo + n] = f2bf(y);
    }
  }
}

// ---------------------------------------------------------------------------
// Fused attention (flash-style, online softmax). One wave per 16-query tile
// of one (b,h). Reads the packed bf16 qkv buffer (rows token*B+b, width 3E;
// q at 0, k at E, v at 2E). Writes bf16 (only consumed by the Wo GEMM).
// D = head dim (64 for E=512, 128 for E=1024).
// ---------------------------------------------------------------------------
template <int D>
__global__ __launch_bounds__(32)
void k_attn(const __bf16* __restrict__ qkv,
            const float* __restrict__ mask,   // (B,T,T) additive or nullptr
            __bf16* __restrict__ outp,        // (T*B, E) rows t*B+b
            int E)
{
  constexpr int NKF = D / 32;   // k-steps over head dim
  constexpr int NCB = D / 16;   // output column blocks
  __shared__ __bf16 plds[16 * 32];

  const int l  = threadIdx.x;
  const int lo = l & 15, hi = l >> 4, kb = hi * 8;
  const int q0 = blockIdx.x * 16;
  const int h  = blockIdx.y;
  const int b  = blockIdx.z;
  const size_t row3E = (size_t)3 * E;

  // Q fragments (held for whole pass)
  v16bf qf[NKF];
  {
    const __bf16* qbase = qkv + ((size_t)(q0 + lo) * BB + b) * row3E + h * D;
    #pragma unroll
    for (int s = 0; s < NKF; ++s)
      qf[s] = CAT16(*(const v8bf*)(qbase + s * 32 + kb),
                    *(const v8bf*)(qbase + s * 32 + kb + 16));
  }

  v8f o[NCB];
  #pragma unroll
  for (int c = 0; c < NCB; ++c) o[c] = (v8f)(0.f);
  float mrow[8], lrow[8];
  #pragma unroll
  for (int i = 0; i < 8; ++i) { mrow[i] = -1e30f; lrow[i] = 0.f; }
  const float scale = rsqrtf((float)D);

  for (int j = 0; j < TT / 32; ++j) {
    const int kbase = j * 32;

    // scores: two 16x16 tiles covering 32 keys
    v8f s0 = (v8f)(0.f), s1 = (v8f)(0.f);
    #pragma unroll
    for (int s = 0; s < NKF; ++s) {
      const __bf16* kp = qkv + ((size_t)(kbase + lo) * BB + b) * row3E + E + h * D + s * 32;
      v16bf kf = CAT16(*(const v8bf*)(kp + kb), *(const v8bf*)(kp + kb + 16));
      s0 = __builtin_amdgcn_wmma_f32_16x16x32_bf16(false, qf[s], false, kf, (short)0, s0, false, false);
    }
    #pragma unroll
    for (int s = 0; s < NKF; ++s) {
      const __bf16* kp = qkv + ((size_t)(kbase + 16 + lo) * BB + b) * row3E + E + h * D + s * 32;
      v16bf kf = CAT16(*(const v8bf*)(kp + kb), *(const v8bf*)(kp + kb + 16));
      s1 = __builtin_amdgcn_wmma_f32_16x16x32_bf16(false, qf[s], false, kf, (short)0, s1, false, false);
    }

    // scale + mask; online softmax update (16-lane butterflies per half)
    float sv0[8], sv1[8];
    #pragma unroll
    for (int i = 0; i < 8; ++i) {
      float a = s0[i] * scale, c2 = s1[i] * scale;
      if (mask) {
        const size_t mb = ((size_t)b * TT + q0 + i + hi * 8) * TT + kbase;
        a  += mask[mb + lo];
        c2 += mask[mb + 16 + lo];
      }
      sv0[i] = a; sv1[i] = c2;
    }
    #pragma unroll
    for (int i = 0; i < 8; ++i) {
      float tmax = fmaxf(sv0[i], sv1[i]);
      #pragma unroll
      for (int off = 1; off < 16; off <<= 1)
        tmax = fmaxf(tmax, __shfl_xor(tmax, off, 32));
      const float nm = fmaxf(mrow[i], tmax);
      const float al = __expf(mrow[i] - nm);
      mrow[i] = nm;
      sv0[i] = __expf(sv0[i] - nm);
      sv1[i] = __expf(sv1[i] - nm);
      float rs = sv0[i] + sv1[i];
      #pragma unroll
      for (int off = 1; off < 16; off <<= 1)
        rs += __shfl_xor(rs, off, 32);
      lrow[i] = lrow[i] * al + rs;
      #pragma unroll
      for (int c = 0; c < NCB; ++c) o[c][i] *= al;
    }

    // P: C-layout -> A-layout via LDS (store bf16 once, re-read with b128)
    __syncthreads();
    #pragma unroll
    for (int i = 0; i < 8; ++i) {
      plds[(i + hi * 8) * 32 + lo]      = f2bf(sv0[i]);
      plds[(i + hi * 8) * 32 + 16 + lo] = f2bf(sv1[i]);
    }
    __syncthreads();
    v16bf pa = CAT16(*(const v8bf*)(plds + lo * 32 + kb),
                     *(const v8bf*)(plds + lo * 32 + 16 + kb));

    // O += P @ V  (V gathered as scalar bf16 across 32 key tokens)
    #pragma unroll
    for (int c = 0; c < NCB; ++c) {
      v16bf vf;
      #pragma unroll
      for (int e = 0; e < 16; ++e) {
        const int kidx = (e < 8) ? (kb + e) : (16 + kb + (e - 8));
        const int kt = kbase + kidx;
        vf[e] = qkv[((size_t)kt * BB + b) * row3E + 2 * (size_t)E + h * D + c * 16 + lo];
      }
      o[c] = __builtin_amdgcn_wmma_f32_16x16x32_bf16(false, pa, false, vf, (short)0, o[c], false, false);
    }
  }

  #pragma unroll
  for (int i = 0; i < 8; ++i) {
    const float inv = 1.f / lrow[i];
    #pragma unroll
    for (int c = 0; c < NCB; ++c) o[c][i] *= inv;
  }
  #pragma unroll
  for (int c = 0; c < NCB; ++c)
    #pragma unroll
    for (int i = 0; i < 8; ++i)
      outp[((size_t)(q0 + i + hi * 8) * BB + b) * E + h * D + c * 16 + lo] = f2bf(o[c][i]);
}

// ---------------------------------------------------------------------------
// Helper kernels
// ---------------------------------------------------------------------------
__global__ void k_layernorm(const float* __restrict__ x, const float* __restrict__ g,
                            const float* __restrict__ bt,
                            float* __restrict__ yf, __bf16* __restrict__ yb,
                            int C, int act)
{
  const int row = blockIdx.x;
  const float* xr = x + (size_t)row * C;
  float s = 0.f, ss = 0.f;
  for (int c = threadIdx.x; c < C; c += blockDim.x) { float v = xr[c]; s += v; ss += v * v; }
  __shared__ float sh1[256], sh2[256];
  sh1[threadIdx.x] = s; sh2[threadIdx.x] = ss; __syncthreads();
  for (int o = 128; o > 0; o >>= 1) {
    if (threadIdx.x < o) { sh1[threadIdx.x] += sh1[threadIdx.x + o]; sh2[threadIdx.x] += sh2[threadIdx.x + o]; }
    __syncthreads();
  }
  const float mean = sh1[0] / C;
  const float var  = sh2[0] / C - mean * mean;
  const float rs   = rsqrtf(var + EPS);
  for (int c = threadIdx.x; c < C; c += blockDim.x) {
    float v = apply_act((xr[c] - mean) * rs * g[c] + bt[c], act);
    if (yf) yf[(size_t)row * C + c] = v;
    if (yb) yb[(size_t)row * C + c] = f2bf(v);
  }
}

__global__ void k_small_linear(const float* __restrict__ A, const float* __restrict__ W,
                               const float* __restrict__ bias, float* __restrict__ out,
                               int M, int N, int K, int act)
{
  const int tid = blockIdx.x * blockDim.x + threadIdx.x;
  if (tid >= M * N) return;
  const int r = tid / N, n = tid % N;
  float s = bias ? bias[n] : 0.f;
  for (int k = 0; k < K; ++k) s += A[(size_t)r * K + k] * W[(size_t)n * K + k];
  out[tid] = apply_act(s, act);
}

__global__ void k_transpose_in(const float* __restrict__ in, __bf16* __restrict__ out)
{ // (B,C_IN,T) -> bf16 rows b*T+t of width 64 (feeds encoder GEMM only)
  const int tid = blockIdx.x * blockDim.x + threadIdx.x;
  if (tid >= MTOT * 64) return;
  const int r = tid >> 6, c = tid & 63;
  const int b = r >> 10, t = r & 1023;
  out[tid] = f2bf(in[((size_t)b * 64 + c) * TT + t]);
}

__global__ void k_concat_enc_hidden(const float* __restrict__ xenc, const float* __restrict__ hidden,
                                    __bf16* __restrict__ out)
{ // bf16 rows b*T+t, width 1024; hidden is (T,B,HC). Feeds addh GEMM only.
  const int tid = blockIdx.x * blockDim.x + threadIdx.x;
  if (tid >= MTOT * 1024) return;
  const int r = tid >> 10, c = tid & 1023;
  const int b = r >> 10, t = r & 1023;
  const float v = (c < HC) ? xenc[(size_t)r * HC + c]
                           : hidden[((size_t)t * BB + b) * HC + (c - HC)];
  out[tid] = f2bf(v);
}

__global__ void k_concat2(const float* __restrict__ A, const float* __restrict__ Bv,
                          float* __restrict__ out)
{ // f32 concat 512+512 -> 1024 (feeds LN + residual)
  const int tid = blockIdx.x * blockDim.x + threadIdx.x;
  if (tid >= MTOT * 1024) return;
  const int r = tid >> 10, c = tid & 1023;
  out[tid] = (c < HC) ? A[(size_t)r * HC + c] : Bv[(size_t)r * HC + (c - HC)];
}

__global__ void k_gate_permute(const float* __restrict__ x, const float* __restrict__ sf,
                               float* __restrict__ out)
{ // rows b*T+t -> rows t*B+b, scaled by sf[row]
  const int tid = blockIdx.x * blockDim.x + threadIdx.x;
  if (tid >= MTOT * HC) return;
  const int r = tid >> 9, c = tid & 511;
  const int b = r >> 10, t = r & 1023;
  out[((size_t)t * BB + b) * HC + c] = x[tid] * sf[r];
}

__global__ void k_mean_tokens(const float* __restrict__ in, float* __restrict__ out,
                              int C, int nrep)
{ // in rows t*B+b -> out[b*C+c], replicated nrep times
  const int tid = blockIdx.x * blockDim.x + threadIdx.x;
  if (tid >= BB * C) return;
  const int b = tid / C, c = tid % C;
  float s = 0.f;
  for (int t = 0; t < TT; ++t) s += in[((size_t)t * BB + b) * C + c];
  s *= (1.f / (float)TT);
  for (int r = 0; r < nrep; ++r) out[(size_t)r * BB * C + tid] = s;
}

__global__ void k_softmax_rows(float* __restrict__ x, int R, int C)
{
  const int r = blockIdx.x * blockDim.x + threadIdx.x;
  if (r >= R) return;
  float mx = -1e30f;
  for (int c = 0; c < C; ++c) mx = fmaxf(mx, x[r * C + c]);
  float s = 0.f;
  for (int c = 0; c < C; ++c) { float e = __expf(x[r * C + c] - mx); x[r * C + c] = e; s += e; }
  const float inv = 1.f / s;
  for (int c = 0; c < C; ++c) x[r * C + c] *= inv;
}

__global__ void k_evw_stats(const float* __restrict__ evw, float* __restrict__ mu,
                            float* __restrict__ rs)
{ // per-slice mean / rsqrt(var+eps) over T*T (biased var, InstanceNorm-like)
  const int k = blockIdx.x;
  const size_t N = (size_t)TT * TT;
  const float* p = evw + (size_t)k * N;
  float s = 0.f, ss = 0.f;
  for (size_t i = threadIdx.x; i < N; i += blockDim.x) { float v = p[i]; s += v; ss += v * v; }
  __shared__ float sh1[256], sh2[256];
  sh1[threadIdx.x] = s; sh2[threadIdx.x] = ss; __syncthreads();
  for (int o = 128; o > 0; o >>= 1) {
    if (threadIdx.x < o) { sh1[threadIdx.x] += sh1[threadIdx.x + o]; sh2[threadIdx.x] += sh2[threadIdx.x + o]; }
    __syncthreads();
  }
  if (threadIdx.x == 0) {
    const float m = sh1[0] / (float)N;
    const float v = sh2[0] / (float)N - m * m;
    mu[k] = m; rs[k] = rsqrtf(v + EPS);
  }
}

__global__ void k_mask_mix(const float* __restrict__ evw, const float* __restrict__ imp,
                           const float* __restrict__ mu, const float* __restrict__ rs,
                           float* __restrict__ mask)
{ // mask[b,t,s] = sum_k imp[b,k] * (evw[k,t,s]-mu[k])*rs[k]
  const size_t tid = (size_t)blockIdx.x * blockDim.x + threadIdx.x;
  const size_t NTS = (size_t)TT * TT;
  if (tid >= (size_t)BB * NTS) return;
  const int b = (int)(tid / NTS);
  const size_t rem = tid % NTS;
  float acc = 0.f;
  #pragma unroll
  for (int k = 0; k < 8; ++k)
    acc += imp[b * 8 + k] * (evw[(size_t)k * NTS + rem] - mu[k]) * rs[k];
  mask[tid] = acc;
}

// ---------------------------------------------------------------------------
// Host orchestration
// ---------------------------------------------------------------------------
static inline dim3 g1d(size_t n, int blk = 256) { return dim3((unsigned)((n + blk - 1) / blk)); }

extern "C" void kernel_launch(void* const* d_in, const int* in_sizes, int n_in,
                              void* d_out, int out_size, void* d_ws, size_t ws_size,
                              hipStream_t stream)
{
  (void)in_sizes; (void)n_in; (void)out_size; (void)ws_size;

  // ---- inputs (setup_inputs dict order, nested dicts flattened) ----
  const float* inp      = (const float*)d_in[0];   // (B, C_IN, H, W)
  const float* hidden   = (const float*)d_in[1];   // (T, B, HC)
  const float* W_enc    = (const float*)d_in[2];
  const float* ln_enc_g = (const float*)d_in[3];
  const float* ln_enc_b = (const float*)d_in[4];
  const float* W_addh   = (const float*)d_in[5];
  const float* b_addh   = (const float*)d_in[6];
  const float* ln_ah_g  = (const float*)d_in[7];
  const float* ln_ah_b  = (const float*)d_in[8];
  const float* W_sf1    = (const float*)d_in[9];
  const float* b_sf1    = (const float*)d_in[10];
  const float* W_sf2    = (const float*)d_in[11];
  const float* b_sf2    = (const float*)d_in[12];
  const float* W_sp1    = (const float*)d_in[13];
  const float* b_sp1    = (const float*)d_in[14];
  const float* W_sp2    = (const float*)d_in[15];
  const float* b_sp2    = (const float*)d_in[16];
  const float* evw      = (const float*)d_in[17];
  const float* ln_m1_g  = (const float*)d_in[18];
  const float* ln_m1_b  = (const float*)d_in[19];
  const float* mWqkv    = (const float*)d_in[20];
  const float* mbqkv    = (const float*)d_in[21];
  const float* mWo      = (const float*)d_in[22];
  const float* mbo      = (const float*)d_in[23];
  const float* mf_lng   = (const float*)d_in[24];
  const float* mf_lnb   = (const float*)d_in[25];
  const float* mf_W1    = (const float*)d_in[26];
  const float* mf_b1    = (const float*)d_in[27];
  const float* mf_W2    = (const float*)d_in[28];
  const float* mf_b2    = (const float*)d_in[29];
  const float* ln_u_g   = (const float*)d_in[30];
  const float* ln_u_b   = (const float*)d_in[31];
  const float* uWqkv    = (const float*)d_in[32];
  const float* ubqkv    = (const float*)d_in[33];
  const float* uWo      = (const float*)d_in[34];
  const float* ubo      = (const float*)d_in[35];
  const float* uf_lng   = (const float*)d_in[36];
  const float* uf_lnb   = (const float*)d_in[37];
  const float* uf_W1    = (const float*)d_in[38];
  const float* uf_b1    = (const float*)d_in[39];
  const float* uf_W2    = (const float*)d_in[40];
  const float* uf_b2    = (const float*)d_in[41];
  const float* W_agg    = (const float*)d_in[42];
  const float* b_agg    = (const float*)d_in[43];
  const float* W_scale  = (const float*)d_in[44];
  const float* b_scale  = (const float*)d_in[45];
  const float* ln_r_g   = (const float*)d_in[46];
  const float* ln_r_b   = (const float*)d_in[47];
  const float* rWqkv    = (const float*)d_in[48];
  const float* rbqkv    = (const float*)d_in[49];
  const float* rWo      = (const float*)d_in[50];
  const float* rbo      = (const float*)d_in[51];
  const float* rf_lng   = (const float*)d_in[52];
  const float* rf_lnb   = (const float*)d_in[53];
  const float* rf_W1    = (const float*)d_in[54];
  const float* rf_b1    = (const float*)d_in[55];
  const float* rf_W2    = (const float*)d_in[56];
  const float* rf_b2    = (const float*)d_in[57];

  float* outp = (float*)d_out;

  // ---- workspace carve-out (bytes, 256B-aligned chunks) ----
  char* base = (char*)d_ws;
  size_t off = 0;
  auto allocB = [&](size_t bytes) -> void* {
    size_t o = (off + 255) & ~(size_t)255;
    off = o + bytes;
    return (void*)(base + o);
  };
  auto fbuf = [&](size_t n) { return (float*)allocB(n * sizeof(float)); };
  auto bbuf = [&](size_t n) { return (__bf16*)allocB(n * sizeof(__bf16)); };

  // bf16 weight copies
  __bf16* wb_enc   = bbuf((size_t)HC * 64);
  __bf16* wb_addh  = bbuf((size_t)HC * 1024);
  __bf16* wb_sf1   = bbuf((size_t)64 * HC);
  __bf16* wb_mqkv  = bbuf((size_t)3 * HC * HC);
  __bf16* wb_mwo   = bbuf((size_t)HC * HC);
  __bf16* wb_mw1   = bbuf((size_t)HC * HC);
  __bf16* wb_mw2   = bbuf((size_t)HC * HC);
  __bf16* wb_uqkv  = bbuf((size_t)3072 * 1024);
  __bf16* wb_uwo   = bbuf((size_t)1024 * 1024);
  __bf16* wb_uw1   = bbuf((size_t)1024 * 1024);
  __bf16* wb_uw2   = bbuf((size_t)1024 * 1024);
  __bf16* wb_agg   = bbuf((size_t)HC * 1024);
  __bf16* wb_scale = bbuf((size_t)HC * HC);
  __bf16* wb_rqkv  = bbuf((size_t)3 * HC * HC);
  __bf16* wb_rwo   = bbuf((size_t)HC * HC);
  __bf16* wb_rw1   = bbuf((size_t)HC * HC);
  __bf16* wb_rw2   = bbuf((size_t)HC * HC);

  // bf16 activations (GEMM A-operands)
  __bf16* xin_b   = bbuf((size_t)MTOT * 64);
  __bf16* cat_b   = bbuf((size_t)MTOT * 1024);
  __bf16* xbuf_b  = bbuf((size_t)MTOT * HC);
  __bf16* t1b     = bbuf((size_t)MTOT * 1024);   // LN bf16 outputs
  __bf16* t2b     = bbuf((size_t)MTOT * 1024);   // FFN hidden
  __bf16* qkvb    = bbuf((size_t)MTOT * 3072);
  __bf16* attnb   = bbuf((size_t)MTOT * 1024);
  __bf16* hv_b    = bbuf((size_t)MTOT * 1024);
  __bf16* newh_b  = bbuf((size_t)MTOT * HC);

  // f32 activations
  float* t1    = fbuf((size_t)MTOT * HC);     // pre-LN GEMM outputs
  float* xenc  = fbuf((size_t)MTOT * HC);
  float* xbuf  = fbuf((size_t)MTOT * HC);
  float* xp    = fbuf((size_t)MTOT * HC);
  float* t2    = fbuf((size_t)MTOT * 64);     // sf hidden
  float* sf    = fbuf((size_t)MTOT);
  float* meanx = fbuf((size_t)BB * HC);
  float* imph  = fbuf((size_t)BB * 64);
  float* imp   = fbuf((size_t)BB * 8);
  float* evmu  = fbuf(8);
  float* evrs  = fbuf(8);
  float* mask  = fbuf((size_t)BB * TT * TT);
  float* mv    = fbuf((size_t)MTOT * HC);
  float* mvin  = fbuf((size_t)MTOT * 1024);
  float* hv    = fbuf((size_t)MTOT * 1024);
  float* outhv = fbuf((size_t)MTOT * HC);
  float* ob    = fbuf((size_t)MTOT * HC);

  const dim3 blk256(256);
  auto gemm_grid = [](int N) { return dim3(MTOT / 128, N / 64); };
  auto cvt = [&](const float* src, __bf16* dst, size_t n) {
    k_cvt_bf16<<<g1d(n / 8), blk256, 0, stream>>>(src, dst, n / 8);
  };

  // 0) one-time bf16 weight conversion (bandwidth-trivial vs ~220 GFLOP)
  cvt(W_enc,   wb_enc,   (size_t)HC * 64);
  cvt(W_addh,  wb_addh,  (size_t)HC * 1024);
  cvt(W_sf1,   wb_sf1,   (size_t)64 * HC);
  cvt(mWqkv,   wb_mqkv,  (size_t)3 * HC * HC);
  cvt(mWo,     wb_mwo,   (size_t)HC * HC);
  cvt(mf_W1,   wb_mw1,   (size_t)HC * HC);
  cvt(mf_W2,   wb_mw2,   (size_t)HC * HC);
  cvt(uWqkv,   wb_uqkv,  (size_t)3072 * 1024);
  cvt(uWo,     wb_uwo,   (size_t)1024 * 1024);
  cvt(uf_W1,   wb_uw1,   (size_t)1024 * 1024);
  cvt(uf_W2,   wb_uw2,   (size_t)1024 * 1024);
  cvt(W_agg,   wb_agg,   (size_t)HC * 1024);
  cvt(W_scale, wb_scale, (size_t)HC * HC);
  cvt(rWqkv,   wb_rqkv,  (size_t)3 * HC * HC);
  cvt(rWo,     wb_rwo,   (size_t)HC * HC);
  cvt(rf_W1,   wb_rw1,   (size_t)HC * HC);
  cvt(rf_W2,   wb_rw2,   (size_t)HC * HC);

  // 1) encoder: relu(LN(x @ W_enc^T))
  k_transpose_in<<<g1d((size_t)MTOT * 64), blk256, 0, stream>>>(inp, xin_b);
  k_gemm<<<gemm_grid(HC), blk256, 0, stream>>>(xin_b, 64, wb_enc, 64, nullptr, nullptr, t1, nullptr, HC, 64, ACT_NONE);
  k_layernorm<<<MTOT, blk256, 0, stream>>>(t1, ln_enc_g, ln_enc_b, xenc, nullptr, HC, ACT_RELU);

  // 2) concat with previous hidden, addh linear + LN (dual out: f32 + bf16)
  k_concat_enc_hidden<<<g1d((size_t)MTOT * 1024), blk256, 0, stream>>>(xenc, hidden, cat_b);
  k_gemm<<<gemm_grid(HC), blk256, 0, stream>>>(cat_b, 1024, wb_addh, 1024, b_addh, nullptr, t1, nullptr, HC, 1024, ACT_NONE);
  k_layernorm<<<MTOT, blk256, 0, stream>>>(t1, ln_ah_g, ln_ah_b, xbuf, xbuf_b, HC, ACT_NONE);

  // 3) spatial-filter gate: sf = sigmoid(W_sf2 @ relu(W_sf1 @ x)); x *= sf, permute
  k_gemm<<<gemm_grid(64), blk256, 0, stream>>>(xbuf_b, HC, wb_sf1, HC, b_sf1, nullptr, t2, nullptr, 64, HC, ACT_RELU);
  k_small_linear<<<g1d((size_t)MTOT), blk256, 0, stream>>>(t2, W_sf2, b_sf2, sf, MTOT, 1, 64, ACT_SIG);
  k_gate_permute<<<g1d((size_t)MTOT * HC), blk256, 0, stream>>>(xbuf, sf, xp);

  // 4) importance: softmax(sigmoid(sp2(relu(sp1(mean_t x)))))
  k_mean_tokens<<<g1d((size_t)BB * HC), blk256, 0, stream>>>(xp, meanx, HC, 1);
  k_small_linear<<<g1d((size_t)BB * 64), blk256, 0, stream>>>(meanx, W_sp1, b_sp1, imph, BB, 64, HC, ACT_RELU);
  k_small_linear<<<g1d((size_t)BB * 8), blk256, 0, stream>>>(imph, W_sp2, b_sp2, imp, BB, 8, 64, ACT_SIG);
  k_softmax_rows<<<1, 32, 0, stream>>>(imp, BB, 8);

  // 5) edge-weight instance norm + per-batch mix -> additive attention mask
  k_evw_stats<<<8, blk256, 0, stream>>>(evw, evmu, evrs);
  k_mask_mix<<<g1d((size_t)BB * TT * TT), blk256, 0, stream>>>(evw, imp, evmu, evrs, mask);

  // 6) m-block: masked MHA (E=512) + FFN, pre-LN, residuals
  k_layernorm<<<MTOT, blk256, 0, stream>>>(xp, ln_m1_g, ln_m1_b, nullptr, t1b, HC, ACT_NONE);
  k_gemm<<<gemm_grid(3 * HC), blk256, 0, stream>>>(t1b, HC, wb_mqkv, HC, mbqkv, nullptr, nullptr, qkvb, 3 * HC, HC, ACT_NONE);
  k_attn<64><<<dim3(TT / 16, NHD, BB), dim3(32), 0, stream>>>(qkvb, mask, attnb, HC);
  k_gemm<<<gemm_grid(HC), blk256, 0, stream>>>(attnb, HC, wb_mwo, HC, mbo, xp, mv, nullptr, HC, HC, ACT_NONE);
  k_layernorm<<<MTOT, blk256, 0, stream>>>(mv, mf_lng, mf_lnb, nullptr, t1b, HC, ACT_NONE);
  k_gemm<<<gemm_grid(HC), blk256, 0, stream>>>(t1b, HC, wb_mw1, HC, mf_b1, nullptr, nullptr, t2b, HC, HC, ACT_GELU);
  k_gemm<<<gemm_grid(HC), blk256, 0, stream>>>(t2b, HC, wb_mw2, HC, mf_b2, mv, mv, nullptr, HC, HC, ACT_NONE);

  // 7) u-block on concat([mv, x]) (E=1024), no mask
  k_concat2<<<g1d((size_t)MTOT * 1024), blk256, 0, stream>>>(mv, xp, mvin);
  k_layernorm<<<MTOT, blk256, 0, stream>>>(mvin, ln_u_g, ln_u_b, nullptr, t1b, 1024, ACT_NONE);
  k_gemm<<<gemm_grid(3072), blk256, 0, stream>>>(t1b, 1024, wb_uqkv, 1024, ubqkv, nullptr, nullptr, qkvb, 3072, 1024, ACT_NONE);
  k_attn<128><<<dim3(TT / 16, NHD, BB), dim3(32), 0, stream>>>(qkvb, nullptr, attnb, 1024);
  k_gemm<<<gemm_grid(1024), blk256, 0, stream>>>(attnb, 1024, wb_uwo, 1024, ubo, mvin, hv, nullptr, 1024, 1024, ACT_NONE);
  k_layernorm<<<MTOT, blk256, 0, stream>>>(hv, uf_lng, uf_lnb, nullptr, t1b, 1024, ACT_NONE);
  k_gemm<<<gemm_grid(1024), blk256, 0, stream>>>(t1b, 1024, wb_uw1, 1024, uf_b1, nullptr, nullptr, t2b, 1024, 1024, ACT_GELU);
  k_gemm<<<gemm_grid(1024), blk256, 0, stream>>>(t2b, 1024, wb_uw2, 1024, uf_b2, hv, hv, hv_b, 1024, 1024, ACT_NONE);

  // 8) aggregate to new hidden (tanh), scale projection
  k_gemm<<<gemm_grid(HC), blk256, 0, stream>>>(hv_b, 1024, wb_agg, 1024, b_agg, nullptr, nullptr, newh_b, HC, 1024, ACT_TANH);
  k_gemm<<<gemm_grid(HC), blk256, 0, stream>>>(newh_b, HC, wb_scale, HC, b_scale, nullptr, outhv, nullptr, HC, HC, ACT_NONE);

  // 9) r-block: MHA (E=512, no mask) + FFN with residuals
  k_layernorm<<<MTOT, blk256, 0, stream>>>(outhv, ln_r_g, ln_r_b, nullptr, t1b, HC, ACT_NONE);
  k_gemm<<<gemm_grid(3 * HC), blk256, 0, stream>>>(t1b, HC, wb_rqkv, HC, rbqkv, nullptr, nullptr, qkvb, 3 * HC, HC, ACT_NONE);
  k_attn<64><<<dim3(TT / 16, NHD, BB), dim3(32), 0, stream>>>(qkvb, nullptr, attnb, HC);
  k_gemm<<<gemm_grid(HC), blk256, 0, stream>>>(attnb, HC, wb_rwo, HC, rbo, outhv, ob, nullptr, HC, HC, ACT_NONE);
  k_layernorm<<<MTOT, blk256, 0, stream>>>(ob, rf_lng, rf_lnb, nullptr, t1b, HC, ACT_NONE);
  k_gemm<<<gemm_grid(HC), blk256, 0, stream>>>(t1b, HC, wb_rw1, HC, rf_b1, nullptr, nullptr, t2b, HC, HC, ACT_GELU);
  k_gemm<<<gemm_grid(HC), blk256, 0, stream>>>(t2b, HC, wb_rw2, HC, rf_b2, ob, ob, nullptr, HC, HC, ACT_NONE);

  // 10) readout: mean over tokens, replicated 3x (reference returns (o,o,o))
  k_mean_tokens<<<g1d((size_t)BB * HC), blk256, 0, stream>>>(ob, outp, HC, 3);
}
